// SpatialAttention_1400159338533
// MI455X (gfx1250) — matrix operand order repaired and checked by reading, not compiled, not run on hardware
//
#include <hip/hip_runtime.h>
#include <hip/hip_bf16.h>
#include <math.h>

#define N_TOK 4096
#define DM    512
#define GAMMA_F 0.5f

typedef __bf16 bf16_t;
typedef __attribute__((ext_vector_type(16))) __bf16 v16bf;
typedef __attribute__((ext_vector_type(8)))  __bf16 v8bf;
typedef __attribute__((ext_vector_type(8)))  float  v8f;

union AFrag { v16bf v; struct { v8bf lo; v8bf hi; } h; };

// A-matrix 16x32 bf16 fragment (M x K), row-major source with leading dim ld.
// lane r=lane&15 holds row r; hi=lane>>4 selects K-offset halves per ISA layout:
// elements 0..7 = K[d0+8*hi .. +7], elements 8..15 = K[d0+16+8*hi .. +7]
__device__ __forceinline__ v16bf load_a_frag_bf16(const bf16_t* base, int ld,
                                                  int r0, int d0, int lane) {
  int r  = lane & 15;
  int hi = lane >> 4;
  const bf16_t* p = base + (size_t)(r0 + r) * ld + d0 + 8 * hi;
  AFrag f;
  f.h.lo = *(const v8bf*)(p);
  f.h.hi = *(const v8bf*)(p + 16);
  return f.v;
}

// B-matrix 32x16 bf16 fragment (K x N), sourced from a row-major matrix whose
// ROWS are the output columns (i.e. B = srcT): lane c=lane&15 holds column c,
// elements 0..7 = K[d0+8*hi .. +7], elements 8..15 = K[d0+16+8*hi .. +7] (matches A).
__device__ __forceinline__ v16bf load_b_frag_bf16(const bf16_t* base, int ld,
                                                  int c0, int d0, int lane) {
  int c  = lane & 15;
  int hi = lane >> 4;
  const bf16_t* p = base + (size_t)(c0 + c) * ld + d0 + 8 * hi;
  AFrag f;
  f.h.lo = *(const v8bf*)(p);
  f.h.hi = *(const v8bf*)(p + 16);
  return f.v;
}

// ---------------- Pass 1: fp32 -> bf16 conversion of x and the 5 weights ----
__global__ void cvt_kernel(const float* __restrict__ x,
                           const float* __restrict__ w0, const float* __restrict__ w1,
                           const float* __restrict__ w2, const float* __restrict__ w3,
                           const float* __restrict__ w4,
                           bf16_t* __restrict__ xb, bf16_t* __restrict__ wb) {
  const int NX = N_TOK * DM;
  const int NW = DM * DM;
  const int total = NX + 5 * NW;
  for (int i = blockIdx.x * blockDim.x + threadIdx.x; i < total;
       i += gridDim.x * blockDim.x) {
    if (i < NX) {
      xb[i] = (bf16_t)x[i];
    } else {
      int j = i - NX;
      int wi = j / NW, off = j % NW;
      const float* w = (wi == 0) ? w0 : (wi == 1) ? w1 : (wi == 2) ? w2
                     : (wi == 3) ? w3 : w4;
      wb[(size_t)wi * NW + off] = (bf16_t)w[off];
    }
  }
}

// ---------------- Pass 2: projections Y = x @ W^T (bf16 WMMA) ----------------
// blockIdx.z selects which of the 5 weight matrices; W_v output is transposed.
__global__ void proj_kernel(const bf16_t* __restrict__ xb, const bf16_t* __restrict__ wb,
                            bf16_t* __restrict__ qr, bf16_t* __restrict__ qf,
                            bf16_t* __restrict__ kr, bf16_t* __restrict__ kf,
                            bf16_t* __restrict__ vT) {
  const int lane = threadIdx.x & 31;
  const int wid  = threadIdx.x >> 5;
  const int c0 = (blockIdx.x * 4 + wid) * 16;
  const int r0 = blockIdx.y * 16;
  const int m  = blockIdx.z;
  const bf16_t* W = wb + (size_t)m * DM * DM;  // row j holds W[j, :] = B column j
  v8f acc = {};
#pragma unroll
  for (int d0 = 0; d0 < DM; d0 += 32) {
    v16bf a = load_a_frag_bf16(xb, DM, r0, d0, lane);
    v16bf b = load_b_frag_bf16(W,  DM, c0, d0, lane);
    acc = __builtin_amdgcn_wmma_f32_16x16x32_bf16(false, a, false, b,
                                                  (short)0, acc, false, false);
  }
  const int n  = lane & 15;
  const int hi = lane >> 4;
  if (m < 4) {
    bf16_t* out = (m == 0) ? qr : (m == 1) ? qf : (m == 2) ? kr : kf;
#pragma unroll
    for (int g = 0; g < 8; ++g) {
      int row = r0 + g + 8 * hi;
      out[(size_t)row * DM + c0 + n] = (bf16_t)acc[g];
    }
  } else {
    v8bf t;
#pragma unroll
    for (int g = 0; g < 8; ++g) t[g] = (bf16_t)acc[g];
    *(v8bf*)(vT + (size_t)(c0 + n) * N_TOK + r0 + 8 * hi) = t;  // vT[col][row..]
  }
}

// ---------------- Pass 3: scores -> exp/mask/combine -> u, rowsums -----------
// One block owns a 16-row block; its 8 waves cover all 64 groups of 4 column
// tiles. A-fragments (Q rows) are loaded once per K-chunk and reused across the
// 4 column tiles (8 WMMAs per 10 loads instead of 2 per 4).
__global__ void scores_kernel(const bf16_t* __restrict__ qr, const bf16_t* __restrict__ kr,
                              const bf16_t* __restrict__ qf, const bf16_t* __restrict__ kf,
                              const unsigned char* __restrict__ maskR,
                              const unsigned char* __restrict__ maskF,
                              float* __restrict__ attn, float* __restrict__ rowsum) {
  __shared__ float part[8][16];
  const int lane = threadIdx.x & 31;
  const int wid  = threadIdx.x >> 5;
  const int r0 = blockIdx.x * 16;
  const int n  = lane & 15;
  const int hi = lane >> 4;
  const float inv_sqrt_d = 0.044194173824159216f;  // 1/sqrt(512)

  float rs[8];
#pragma unroll
  for (int g = 0; g < 8; ++g) rs[g] = 0.0f;

  for (int ctg = wid; ctg < N_TOK / 64; ctg += 8) {
    const int cbase = ctg * 64;
    v8f accR[4] = {};
    v8f accF[4] = {};
#pragma unroll
    for (int d0 = 0; d0 < DM; d0 += 32) {
      v16bf aR = load_a_frag_bf16(qr, DM, r0, d0, lane);
      v16bf aF = load_a_frag_bf16(qf, DM, r0, d0, lane);
#pragma unroll
      for (int t = 0; t < 4; ++t) {
        v16bf bR = load_b_frag_bf16(kr, DM, cbase + 16 * t, d0, lane);
        accR[t] = __builtin_amdgcn_wmma_f32_16x16x32_bf16(false, aR, false, bR,
                                                          (short)0, accR[t], false, false);
        v16bf bF = load_b_frag_bf16(kf, DM, cbase + 16 * t, d0, lane);
        accF[t] = __builtin_amdgcn_wmma_f32_16x16x32_bf16(false, aF, false, bF,
                                                          (short)0, accF[t], false, false);
      }
    }
#pragma unroll
    for (int t = 0; t < 4; ++t) {
      const int c0 = cbase + 16 * t;
#pragma unroll
      for (int g = 0; g < 8; ++g) {
        int row = r0 + g + 8 * hi;
        size_t idx = (size_t)row * N_TOK + c0 + n;
        float u = 0.0f;
        if (!maskR[idx]) u += __expf(accR[t][g] * inv_sqrt_d);
        if (!maskF[idx]) u += GAMMA_F * __expf(accF[t][g] * inv_sqrt_d);
        attn[idx] = u;          // unnormalized combined score
        rs[g] += u;
      }
    }
  }

  // deterministic reduction: within 16-lane halves, then fixed-order over waves
#pragma unroll
  for (int g = 0; g < 8; ++g) {
    float v = rs[g];
    v += __shfl_xor(v, 8, 32);
    v += __shfl_xor(v, 4, 32);
    v += __shfl_xor(v, 2, 32);
    v += __shfl_xor(v, 1, 32);
    rs[g] = v;                 // lane 0 (rows 0-7) / lane 16 (rows 8-15) hold sums
  }
  if (n == 0) {
#pragma unroll
    for (int g = 0; g < 8; ++g) part[wid][g + 8 * hi] = rs[g];
  }
  __syncthreads();
  if (threadIdx.x < 16) {
    float s = 0.0f;
#pragma unroll
    for (int w = 0; w < 8; ++w) s += part[w][threadIdx.x];
    rowsum[r0 + threadIdx.x] = s;
  }
}

// ---------------- Pass 4: normalize attn in place ----------------------------
__global__ void norm_kernel(float* __restrict__ attn, const float* __restrict__ rowsum) {
  const size_t total = (size_t)N_TOK * N_TOK / 4;   // float4 elements, 1024/row
  float4* p = (float4*)attn;
  for (size_t i = (size_t)blockIdx.x * blockDim.x + threadIdx.x; i < total;
       i += (size_t)gridDim.x * blockDim.x) {
    int row = (int)(i >> 10);
    float inv = 1.0f / rowsum[row];
    float4 v = p[i];
    v.x *= inv; v.y *= inv; v.z *= inv; v.w *= inv;
    p[i] = v;
  }
}

// ---------------- Pass 5: cntx = attn @ V (bf16 WMMA, A converted on the fly)
__global__ void ctx_kernel(const float* __restrict__ attn, const bf16_t* __restrict__ vT,
                           float* __restrict__ cntx) {
  const int lane = threadIdx.x & 31;
  const int wid  = threadIdx.x >> 5;
  const int r0 = blockIdx.x * 16;
  const int r  = lane & 15;
  const int hi = lane >> 4;
  const int cbase = wid * 64;                 // 4 cntx column tiles per wave
  v8f acc0 = {}, acc1 = {}, acc2 = {}, acc3 = {};
  const float* arow = attn + (size_t)(r0 + r) * N_TOK;

  for (int j0 = 0; j0 < N_TOK; j0 += 32) {
    const float4* pa = (const float4*)(arow + j0 + 8 * hi);
    float4 f0 = pa[0];
    float4 f1 = pa[1];
    float4 f2 = pa[4];   // +16 floats
    float4 f3 = pa[5];
    AFrag af;
    af.h.lo[0] = (bf16_t)f0.x; af.h.lo[1] = (bf16_t)f0.y;
    af.h.lo[2] = (bf16_t)f0.z; af.h.lo[3] = (bf16_t)f0.w;
    af.h.lo[4] = (bf16_t)f1.x; af.h.lo[5] = (bf16_t)f1.y;
    af.h.lo[6] = (bf16_t)f1.z; af.h.lo[7] = (bf16_t)f1.w;
    af.h.hi[0] = (bf16_t)f2.x; af.h.hi[1] = (bf16_t)f2.y;
    af.h.hi[2] = (bf16_t)f2.z; af.h.hi[3] = (bf16_t)f2.w;
    af.h.hi[4] = (bf16_t)f3.x; af.h.hi[5] = (bf16_t)f3.y;
    af.h.hi[6] = (bf16_t)f3.z; af.h.hi[7] = (bf16_t)f3.w;

    v16bf b0 = load_b_frag_bf16(vT, N_TOK, cbase,      j0, lane);
    v16bf b1 = load_b_frag_bf16(vT, N_TOK, cbase + 16, j0, lane);
    v16bf b2 = load_b_frag_bf16(vT, N_TOK, cbase + 32, j0, lane);
    v16bf b3 = load_b_frag_bf16(vT, N_TOK, cbase + 48, j0, lane);
    acc0 = __builtin_amdgcn_wmma_f32_16x16x32_bf16(false, af.v, false, b0, (short)0, acc0, false, false);
    acc1 = __builtin_amdgcn_wmma_f32_16x16x32_bf16(false, af.v, false, b1, (short)0, acc1, false, false);
    acc2 = __builtin_amdgcn_wmma_f32_16x16x32_bf16(false, af.v, false, b2, (short)0, acc2, false, false);
    acc3 = __builtin_amdgcn_wmma_f32_16x16x32_bf16(false, af.v, false, b3, (short)0, acc3, false, false);
  }
  const int n = lane & 15;
#pragma unroll
  for (int g = 0; g < 8; ++g) {
    int row = r0 + g + 8 * hi;
    cntx[(size_t)row * DM + cbase +      n] = acc0[g];
    cntx[(size_t)row * DM + cbase + 16 + n] = acc1[g];
    cntx[(size_t)row * DM + cbase + 32 + n] = acc2[g];
    cntx[(size_t)row * DM + cbase + 48 + n] = acc3[g];
  }
}

extern "C" void kernel_launch(void* const* d_in, const int* in_sizes, int n_in,
                              void* d_out, int out_size, void* d_ws, size_t ws_size,
                              hipStream_t stream) {
  (void)in_sizes; (void)n_in; (void)out_size; (void)ws_size;
  const float* x  = (const float*)d_in[0];
  const unsigned char* maskR = (const unsigned char*)d_in[1];
  const unsigned char* maskF = (const unsigned char*)d_in[2];
  const float* wqr = (const float*)d_in[3];
  const float* wqf = (const float*)d_in[4];
  const float* wkr = (const float*)d_in[5];
  const float* wkf = (const float*)d_in[6];
  const float* wv  = (const float*)d_in[7];

  float* out  = (float*)d_out;
  float* cntx = out;                                  // [4096, 512]
  float* attn = out + (size_t)N_TOK * DM;             // [4096, 4096]

  // workspace carve-up (bf16 staging + rowsums), ~27 MB
  char* w = (char*)d_ws;
  const size_t SZ_XB = (size_t)N_TOK * DM * sizeof(bf16_t);   // 4 MB
  const size_t SZ_WB = (size_t)5 * DM * DM * sizeof(bf16_t);  // 2.5 MB
  const size_t SZ_QK = (size_t)N_TOK * DM * sizeof(bf16_t);   // 4 MB each
  bf16_t* xb = (bf16_t*)w;            w += SZ_XB;
  bf16_t* wb = (bf16_t*)w;            w += SZ_WB;
  bf16_t* qr = (bf16_t*)w;            w += SZ_QK;
  bf16_t* qf = (bf16_t*)w;            w += SZ_QK;
  bf16_t* kr = (bf16_t*)w;            w += SZ_QK;
  bf16_t* kf = (bf16_t*)w;            w += SZ_QK;
  bf16_t* vT = (bf16_t*)w;            w += SZ_QK;     // [512, 4096] transposed
  float* rowsum = (float*)w;          w += (size_t)N_TOK * sizeof(float);

  cvt_kernel<<<2048, 256, 0, stream>>>(x, wqr, wqf, wkr, wkf, wv, xb, wb);

  dim3 pg(8, N_TOK / 16, 5);
  proj_kernel<<<pg, 128, 0, stream>>>(xb, wb, qr, qf, kr, kf, vT);

  scores_kernel<<<N_TOK / 16, 256, 0, stream>>>(qr, kr, qf, kf, maskR, maskF,
                                                attn, rowsum);

  norm_kernel<<<4096, 256, 0, stream>>>(attn, rowsum);

  ctx_kernel<<<N_TOK / 16, 256, 0, stream>>>(attn, vT, cntx);
}
